// Attention_51986284151356
// MI455X (gfx1250) — matrix operand order repaired and checked by reading, not compiled, not run on hardware
//
#include <hip/hip_runtime.h>
#include <hip/hip_bf16.h>
#include <stdint.h>

// ---------------------------------------------------------------------------
// Types for CDNA5 WMMA (wave32): 16x16x32 bf16 -> f32
// ---------------------------------------------------------------------------
typedef __attribute__((ext_vector_type(16))) __bf16 v16bf;
typedef __attribute__((ext_vector_type(8)))  float  v8f;
typedef unsigned short u16;

union AF {
    v16bf v;
    unsigned int u[8];
};

__device__ __forceinline__ v8f vzero() {
    v8f z = {0.f, 0.f, 0.f, 0.f, 0.f, 0.f, 0.f, 0.f};
    return z;
}

__device__ __forceinline__ u16 f2bf(float f) {
    unsigned int u = __builtin_bit_cast(unsigned int, f);
    u += 0x7FFFu + ((u >> 16) & 1u);   // round-to-nearest-even
    return (u16)(u >> 16);
}

__device__ __forceinline__ float halfmax16(float v) {
    // reduce across the 16 lanes of this lane's half (xor of bits 0..3 stays in half)
    v = fmaxf(v, __shfl_xor(v, 1, 32));
    v = fmaxf(v, __shfl_xor(v, 2, 32));
    v = fmaxf(v, __shfl_xor(v, 4, 32));
    v = fmaxf(v, __shfl_xor(v, 8, 32));
    return v;
}

__device__ __forceinline__ float halfsum16(float v) {
    v += __shfl_xor(v, 1, 32);
    v += __shfl_xor(v, 2, 32);
    v += __shfl_xor(v, 4, 32);
    v += __shfl_xor(v, 8, 32);
    return v;
}

#define WMMA_BF16(a, b, c) \
    __builtin_amdgcn_wmma_f32_16x16x32_bf16(false, (a), false, (b), (short)0, (c), false, false)

// ---------------------------------------------------------------------------
// Problem constants
// ---------------------------------------------------------------------------
#define BATCH 2
#define SEQ   2048
#define DIM   1024
#define HEADS 16
#define DHEAD 64
#define QKVN  3072
#define MTOT  (BATCH * SEQ)          // 4096
#define SCALEQ 0.125f                // 64^-0.5
#define LOG2E 1.4426950408889634f
#define NEGBIG (-3.0e38f)

// ---------------------------------------------------------------------------
// Kernel 1: f32 -> bf16 conversion
// ---------------------------------------------------------------------------
__global__ void cvt_bf16_kernel(const float* __restrict__ in, u16* __restrict__ out, int n) {
    int i = blockIdx.x * blockDim.x + threadIdx.x;
    if (i < n) out[i] = f2bf(in[i]);
}

// ---------------------------------------------------------------------------
// Kernel 2/5: bf16 WMMA GEMM.  C[M,N] = A[M,K] * W[N,K]^T (+ bias)
// 128 threads = 4 waves; workgroup tile 128x64, each wave 32 rows x 64 cols.
// Double-buffered fragments: loads for K-step k+1 issue before WMMAs of step k.
// ---------------------------------------------------------------------------
template <int K>
__global__ __launch_bounds__(128) void gemm_bf16_kernel(
    const u16* __restrict__ A, const u16* __restrict__ W,
    float* __restrict__ C, const float* __restrict__ bias,
    int M, int N)
{
    const int wave = threadIdx.x >> 5;
    const int lane = threadIdx.x & 31;
    const int hi   = lane >> 4;
    const int lr   = lane & 15;

    const int row0 = blockIdx.y * 128 + wave * 32;
    const int col0 = blockIdx.x * 64;

    v8f acc[2][4];
#pragma unroll
    for (int m = 0; m < 2; ++m)
#pragma unroll
        for (int t = 0; t < 4; ++t) acc[m][t] = vzero();

    const u16* Arow0 = A + (size_t)(row0 + lr) * K;
    const u16* Arow1 = A + (size_t)(row0 + 16 + lr) * K;

    AF a[2][2];      // [stage][row tile]
    AF b[2][4];      // [stage][col tile]

    auto loadA = [&](AF* dst, int k0) {
#pragma unroll
        for (int p = 0; p < 8; ++p) {
            const int k = 2 * p + 16 * hi;   // A 16x32 bf16 layout
            dst[0].u[p] = *(const unsigned int*)(Arow0 + k0 + k);
            dst[1].u[p] = *(const unsigned int*)(Arow1 + k0 + k);
        }
    };
    auto loadB = [&](AF* dst, int k0) {
#pragma unroll
        for (int t = 0; t < 4; ++t) {
            const u16* Wrow = W + (size_t)(col0 + 16 * t + lr) * K + k0 + 16 * hi;
#pragma unroll
            for (int p = 0; p < 8; ++p)                          // B 32x16 bf16 layout
                dst[t].u[p] = *(const unsigned int*)(Wrow + 2 * p);
        }
    };

    loadA(a[0], 0);
    loadB(b[0], 0);

#pragma unroll 2
    for (int k0 = 0; k0 < K; k0 += 32) {
        const int cur = (k0 >> 5) & 1;
        const int nxt = cur ^ 1;
        if (k0 + 32 < K) {
            loadA(a[nxt], k0 + 32);
            loadB(b[nxt], k0 + 32);
        }
#pragma unroll
        for (int t = 0; t < 4; ++t) {
            acc[0][t] = WMMA_BF16(a[cur][0].v, b[cur][t].v, acc[0][t]);
            acc[1][t] = WMMA_BF16(a[cur][1].v, b[cur][t].v, acc[1][t]);
        }
    }

#pragma unroll
    for (int t = 0; t < 4; ++t) {
        const int col = col0 + 16 * t + lr;
        const float bv = bias ? bias[col] : 0.0f;
#pragma unroll
        for (int m = 0; m < 2; ++m)
#pragma unroll
            for (int r = 0; r < 8; ++r) {
                const int row = row0 + 16 * m + r + 8 * hi;       // C/D layout
                C[(size_t)row * N + col] = acc[m][t][r] + bv;
            }
    }
}

// ---------------------------------------------------------------------------
// Kernel 3: rotary embedding (applied to q,k,v per reference), q *= scale.
// Q,K stored per-head row-major [b][h][n][64]; V stored TRANSPOSED per head
// [b][h][d][n] so attention PV B-fragments are contiguous along keys.
// ---------------------------------------------------------------------------
__global__ void rotary_kernel(const float* __restrict__ qkv, const float* __restrict__ rope,
                              u16* __restrict__ Q, u16* __restrict__ K, u16* __restrict__ Vt)
{
    const int idx = blockIdx.x * blockDim.x + threadIdx.x;
    if (idx >= MTOT * QKVN) return;
    const int m = idx / QKVN;            // b*SEQ + n
    const int c = idx - m * QKVN;
    const int which = c >> 10;           // 0=q,1=k,2=v
    const int rem = c & 1023;
    const int h = rem >> 6;
    const int d = rem & 63;
    const int b = m >> 11;
    const int n = m & 2047;

    const float t  = qkv[(size_t)m * QKVN + c];
    const float tp = qkv[(size_t)m * QKVN + (c ^ 1)];  // rotate_half partner
    const float f  = rope[n * DHEAD + d];
    float r = t * cosf(f) + ((d & 1) ? tp : -tp) * sinf(f);
    if (which == 0) r *= SCALEQ;

    const u16 v = f2bf(r);
    const int bh = b * HEADS + h;
    if (which == 0)      Q[((size_t)bh * SEQ + n) * DHEAD + d] = v;
    else if (which == 1) K[((size_t)bh * SEQ + n) * DHEAD + d] = v;
    else                 Vt[((size_t)bh * DHEAD + d) * SEQ + n] = v;   // transposed
}

// ---------------------------------------------------------------------------
// Kernel 4: causal flash attention. One wave per 16-query tile; key blocks of
// 32.  K fragments: packed b128 loads (K row-major).  V fragments: packed
// loads from transposed V.  Next block prefetched via global_prefetch.
// P transposed C-layout -> A-layout through LDS.
// Output written as bf16 [b][n][h*64] for the final projection GEMM.
// ---------------------------------------------------------------------------
__global__ __launch_bounds__(128) void attn_kernel(
    const u16* __restrict__ Q, const u16* __restrict__ K, const u16* __restrict__ Vt,
    const unsigned char* __restrict__ mask, u16* __restrict__ O)
{
    __shared__ __attribute__((aligned(16))) u16 pbuf[4 * 16 * 32];

    const int wave = threadIdx.x >> 5;
    const int lane = threadIdx.x & 31;
    const int hi   = lane >> 4;
    const int lr   = lane & 15;

    const int gt = blockIdx.x * 4 + wave;     // global 16-query tile id
    const int bh = gt >> 7;                   // 128 tiles per (b,h)
    const int q0 = (gt & 127) << 4;
    const int b  = bh >> 4;
    const int h  = bh & 15;

    const u16* Qh = Q  + (size_t)bh * (SEQ * DHEAD);
    const u16* Kh = K  + (size_t)bh * (SEQ * DHEAD);
    const u16* Vh = Vt + (size_t)bh * (SEQ * DHEAD);   // [d][n]
    const unsigned char* mb = mask + b * SEQ;
    u16* pb = pbuf + wave * (16 * 32);

    // Q A-fragments for the two K-steps (d=0..31, d=32..63)
    AF aq0, aq1;
    {
        const u16* qrow = Qh + (size_t)(q0 + lr) * DHEAD;
#pragma unroll
        for (int p = 0; p < 8; ++p) {
            const int k = 2 * p + 16 * hi;
            aq0.u[p] = *(const unsigned int*)(qrow + k);
            aq1.u[p] = *(const unsigned int*)(qrow + 32 + k);
        }
    }

    float mrow[8], lrow[8];
    v8f o[4];
#pragma unroll
    for (int r = 0; r < 8; ++r) { mrow[r] = NEGBIG; lrow[r] = 0.f; }
#pragma unroll
    for (int t = 0; t < 4; ++t) o[t] = vzero();

    const int nblk = (q0 + 15) / 32 + 1;      // causal bound
    for (int jb = 0; jb < nblk; ++jb) {
        const int j0 = jb * 32;

        // ---- prefetch next key/value block (global_prefetch) ----
        if (jb + 1 < nblk) {
            __builtin_prefetch(Kh + (size_t)(j0 + 32 + lane) * DHEAD, 0, 3);
            __builtin_prefetch(Vh + (size_t)lane * SEQ + j0 + 32, 0, 3);
            __builtin_prefetch(Vh + (size_t)(lane + 32) * SEQ + j0 + 32, 0, 3);
        }

        // ---- load all 4 K B-fragments, then S = Q * K^T ----
        AF bk[4];  // [g*2+s]: column group g (keys j0+16g..), K-step s (d 32s..)
        {
            const u16* kr0 = Kh + (size_t)(j0 + lr) * DHEAD + 16 * hi;
            const u16* kr1 = Kh + (size_t)(j0 + 16 + lr) * DHEAD + 16 * hi;
#pragma unroll
            for (int p = 0; p < 8; ++p) {
                bk[0].u[p] = *(const unsigned int*)(kr0 + 2 * p);
                bk[1].u[p] = *(const unsigned int*)(kr0 + 32 + 2 * p);
                bk[2].u[p] = *(const unsigned int*)(kr1 + 2 * p);
                bk[3].u[p] = *(const unsigned int*)(kr1 + 32 + 2 * p);
            }
        }
        v8f s0 = vzero(), s1 = vzero();
        s0 = WMMA_BF16(aq0.v, bk[0].v, s0);
        s0 = WMMA_BF16(aq1.v, bk[1].v, s0);
        s1 = WMMA_BF16(aq0.v, bk[2].v, s1);
        s1 = WMMA_BF16(aq1.v, bk[3].v, s1);

        // ---- load V B-fragments early from transposed V (packed along keys) ----
        AF bv[4];
#pragma unroll
        for (int t = 0; t < 4; ++t) {
            const u16* vr = Vh + (size_t)(16 * t + lr) * SEQ + j0 + 16 * hi;
#pragma unroll
            for (int p = 0; p < 8; ++p)
                bv[t].u[p] = *(const unsigned int*)(vr + 2 * p);
        }

        // ---- masking + online softmax ----
        float cf[8];
#pragma unroll
        for (int r = 0; r < 8; ++r) {
            const int rowg = q0 + r + 8 * hi;
            const int c0 = j0 + lr, c1 = j0 + 16 + lr;
            float sv0 = s0[r], sv1 = s1[r];
            if (c0 > rowg || !mb[c0]) sv0 = NEGBIG;
            if (c1 > rowg || !mb[c1]) sv1 = NEGBIG;
            const float mx = halfmax16(fmaxf(sv0, sv1));
            const float mn = fmaxf(mrow[r], mx);
            const float c  = exp2f((mrow[r] - mn) * LOG2E);
            const float p0 = exp2f((sv0 - mn) * LOG2E);
            const float p1 = exp2f((sv1 - mn) * LOG2E);
            mrow[r] = mn;
            cf[r]   = c;
            lrow[r] = lrow[r] * c + p0 + p1;
            s0[r] = p0;
            s1[r] = p1;
        }
#pragma unroll
        for (int t = 0; t < 4; ++t)
#pragma unroll
            for (int r = 0; r < 8; ++r) o[t][r] *= cf[r];

        // ---- transpose P (C-layout -> A-layout) through LDS ----
#pragma unroll
        for (int r = 0; r < 8; ++r) {
            const int row = r + 8 * hi;
            pb[row * 32 + lr]      = f2bf(s0[r]);
            pb[row * 32 + 16 + lr] = f2bf(s1[r]);
        }
        asm volatile("s_wait_dscnt 0" ::: "memory");   // per-wave LDS RAW fence
        AF ap;
#pragma unroll
        for (int p = 0; p < 8; ++p) {
            const int k = 2 * p + 16 * hi;
            ap.u[p] = *(const unsigned int*)(pb + lr * 32 + k);
        }

        // ---- O += P * V ----
#pragma unroll
        for (int t = 0; t < 4; ++t)
            o[t] = WMMA_BF16(ap.v, bv[t].v, o[t]);
    }

    // ---- normalize and store bf16 [b][n][h*64] ----
    float inv[8];
#pragma unroll
    for (int r = 0; r < 8; ++r) inv[r] = 1.0f / halfsum16(lrow[r]);

#pragma unroll
    for (int t = 0; t < 4; ++t)
#pragma unroll
        for (int r = 0; r < 8; ++r) {
            const int rowg = q0 + r + 8 * hi;
            const int col  = h * DHEAD + 16 * t + lr;
            O[((size_t)b * SEQ + rowg) * DIM + col] = f2bf(o[t][r] * inv[r]);
        }
}

// ---------------------------------------------------------------------------
// Host-side launch
// ---------------------------------------------------------------------------
extern "C" void kernel_launch(void* const* d_in, const int* in_sizes, int n_in,
                              void* d_out, int out_size, void* d_ws, size_t ws_size,
                              hipStream_t stream) {
    const float*         x     = (const float*)d_in[0];
    const unsigned char* mask  = (const unsigned char*)d_in[1];
    const float*         rope  = (const float*)d_in[2];
    const float*         w_qkv = (const float*)d_in[3];
    const float*         w_out = (const float*)d_in[4];
    const float*         b_out = (const float*)d_in[5];
    float*               out   = (float*)d_out;

    char* ws = (char*)d_ws;
    // Workspace layout (bytes)
    u16*   x_bf   = (u16*)(ws + 0);                         //  8,388,608
    u16*   wq_bf  = (u16*)(ws + 8388608);                   //  6,291,456
    u16*   wo_bf  = (u16*)(ws + 14680064);                  //  2,097,152
    float* qkv    = (float*)(ws + 16777216);                // 50,331,648
    u16*   Qb     = (u16*)(ws + 67108864);                  //  8,388,608
    u16*   Kb     = (u16*)(ws + 75497472);                  //  8,388,608
    u16*   Vtb    = (u16*)(ws + 83886080);                  //  8,388,608  (transposed)
    u16*   attn   = (u16*)(ws + 92274688);                  //  8,388,608  (total ~96MB)

    // 1. convert inputs to bf16
    cvt_bf16_kernel<<<(MTOT * DIM + 255) / 256, 256, 0, stream>>>(x, x_bf, MTOT * DIM);
    cvt_bf16_kernel<<<(QKVN * DIM + 255) / 256, 256, 0, stream>>>(w_qkv, wq_bf, QKVN * DIM);
    cvt_bf16_kernel<<<(DIM * DIM + 255) / 256, 256, 0, stream>>>(w_out, wo_bf, DIM * DIM);

    // 2. QKV projection: [4096,1024] x [3072,1024]^T -> f32 [4096,3072]
    {
        dim3 grid(QKVN / 64, MTOT / 128);
        gemm_bf16_kernel<DIM><<<grid, 128, 0, stream>>>(x_bf, wq_bf, qkv, nullptr, MTOT, QKVN);
    }

    // 3. rotary + scale + per-head layout (V transposed)
    rotary_kernel<<<(MTOT * QKVN + 255) / 256, 256, 0, stream>>>(qkv, rope, Qb, Kb, Vtb);

    // 4. causal flash attention: 4096 query tiles, 4 waves per block
    attn_kernel<<<(BATCH * HEADS * (SEQ / 16)) / 4, 128, 0, stream>>>(Qb, Kb, Vtb, mask, attn);

    // 5. output projection + bias: [4096,1024] x [1024,1024]^T -> f32 d_out
    {
        dim3 grid(DIM / 64, MTOT / 128);
        gemm_bf16_kernel<DIM><<<grid, 128, 0, stream>>>(attn, wo_bf, out, b_out, MTOT, DIM);
    }
}
